// ScaleHierarchicalOptimizer_16664473108648
// MI455X (gfx1250) — matrix operand order repaired and checked by reading, not compiled, not run on hardware
//
#include <hip/hip_runtime.h>
#include <hip/hip_bf16.h>

typedef __attribute__((ext_vector_type(16))) _Float16 v16h;
typedef __attribute__((ext_vector_type(8)))  _Float16 v8h;
typedef __attribute__((ext_vector_type(8)))  float    v8f;

#define T_SCALES 4
#define P_UNITS 10
#define T_UNITS 5
#define G_UNITS 5
#define P_IN 19   // 2*T_SCALES + 1 + T_UNITS + G_UNITS
#define T_IN 24   // 2*T_SCALES + 1 + P_UNITS + G_UNITS

// ---------------------------------------------------------------------------
// Workspace layout:
//   floats [0]=sum d_theta^2, [1]=sum lr,
//   [2..5]=sum m, [6..9]=sum gamma_feat, [10]=sum nu_rel, [11..20]=sum h_new
//   floats [256 ...]: packed f16 B-fragments, 128 bytes per lane x 32 lanes
//       per lane: 4 fragments (bx_lo, bx_hi, bh_lo, bh_hi) x 16 halves
// ---------------------------------------------------------------------------
#define WS_FRAG_OFF 256

__global__ void opt_init(float* ws) {
    if (threadIdx.x < 32) ws[threadIdx.x] = 0.f;
}

// ---------------------------------------------------------------------------
// Prep: one wave builds the per-lane WMMA B fragments for pk (19x30) and
// prk (10x30), K padded to 32, N padded to 32, packed f16, stored to ws.
// B k order used here (32x16 f16, same as A): lane holds column (lane&15); halves
// 0..15 hold K = 16*(lane>>4) + 0..15.
// ---------------------------------------------------------------------------
__global__ void opt_prep(const float* __restrict__ pk,
                         const float* __restrict__ prk,
                         float* ws)
{
    int lane = threadIdx.x;
    if (lane >= 32) return;
    int row = lane & 15;
    int hi  = lane >> 4;
    _Float16 frag[4][16];
#pragma unroll
    for (int j = 0; j < 16; ++j) {
        int K  = 16 * hi + j;
        int cl = row;
        int ch = 16 + row;
        frag[0][j] = (_Float16)((K < P_IN)               ? pk[K * 30 + cl]  : 0.f);
        frag[1][j] = (_Float16)((K < P_IN    && ch < 30) ? pk[K * 30 + ch]  : 0.f);
        frag[2][j] = (_Float16)((K < P_UNITS)            ? prk[K * 30 + cl] : 0.f);
        frag[3][j] = (_Float16)((K < P_UNITS && ch < 30) ? prk[K * 30 + ch] : 0.f);
    }
    char* dst = (char*)(ws + WS_FRAG_OFF) + lane * 128;
#pragma unroll
    for (int f = 0; f < 4; ++f)
#pragma unroll
        for (int c = 0; c < 2; ++c) {
            v8h v;
#pragma unroll
            for (int j = 0; j < 8; ++j) v[j] = frag[f][c * 8 + j];
            *(v8h*)(dst + f * 32 + c * 16) = v;
        }
}

// ---------------------------------------------------------------------------
// Pass A: elementwise EMA updates + global reductions (||d_theta||^2, sum lr)
// ---------------------------------------------------------------------------
__global__ void opt_passA(
    const float* __restrict__ grads, const float* __restrict__ g_bar,
    const float* __restrict__ lam, const float* __restrict__ h_param,
    const float* __restrict__ nu_bar, const float* __restrict__ lr,
    const float* __restrict__ w_dt, const float* __restrict__ b_dt,
    const float* __restrict__ w_dn, const float* __restrict__ b_dn,
    const float* __restrict__ w_bg, const float* __restrict__ b_bg,
    const float* __restrict__ w_bl, const float* __restrict__ b_bl,
    const float* __restrict__ gamma_p,
    float* __restrict__ o_gbar, float* __restrict__ o_lam,
    float* __restrict__ o_nu, float* __restrict__ o_nubar,
    float* ws, int n)
{
    __shared__ float s_acc[2];
    if (threadIdx.x < 2) s_acc[threadIdx.x] = 0.f;
    __syncthreads();

    int idx = blockIdx.x * blockDim.x + threadIdx.x;
    float dt2 = 0.f, lrv = 0.f;
    if (idx < n) {
        float h[P_UNITS];
#pragma unroll
        for (int u = 0; u < P_UNITS; ++u) h[u] = h_param[(size_t)idx * P_UNITS + u];

        float abg = b_bg[0], abl = b_bl[0], adt = b_dt[0], adn = b_dn[0];
#pragma unroll
        for (int u = 0; u < P_UNITS; ++u) {
            abg += h[u] * w_bg[u];
            abl += h[u] * w_bl[u];
            adt += h[u] * w_dt[u];
            adn += h[u] * w_dn[u];
        }
        float b1 = 1.f / (1.f + expf(-abg));   // beta_g
        float b2 = 1.f / (1.f + expf(-abl));   // beta_l

        float g  = grads[idx];
        float g2 = g * g;
#pragma unroll
        for (int s = 0; s < T_SCALES; ++s) {
            if (s) { b1 = sqrtf(b1); b2 = sqrtf(b2); }   // beta ** (0.5 ** s)
            size_t o = (size_t)s * n + idx;
            o_gbar[o] = b1 * g_bar[o] + (1.f - b1) * g;
            o_lam[o]  = b2 * lam[o]  + (1.f - b2) * g2;
        }
        float nb = nu_bar[idx];
        float nu = adn + nb;
        o_nu[idx] = nu;
        float gm = gamma_p[0];
        o_nubar[idx] = gm * nb + (1.f - gm) * nu;

        dt2 = adt * adt;    // d_theta^2
        lrv = lr[idx];
    }
    atomicAdd(&s_acc[0], dt2);
    atomicAdd(&s_acc[1], lrv);
    __syncthreads();
    if (threadIdx.x == 0) {
        atomicAdd(&ws[0], s_acc[0]);
        atomicAdd(&ws[1], s_acc[1]);
    }
}

// ---------------------------------------------------------------------------
// Pass B: per-parameter GRU via WMMA f16 16x16x32, delta_theta, feature sums.
// Block = 256 threads = 8 wave32s; block handles 128 elements (16 per wave).
// ---------------------------------------------------------------------------
__global__ void opt_passB(
    const float* __restrict__ gbar_new, const float* __restrict__ lam_new,
    const float* __restrict__ h_param, const float* __restrict__ lr,
    const float* __restrict__ h_tensor, const float* __restrict__ h_global,
    const float* __restrict__ pb,
    const float* __restrict__ w_dt, const float* __restrict__ b_dt,
    float* __restrict__ o_delta, float* __restrict__ o_hparam,
    float* ws, int n)
{
    __shared__ _Float16 xh_s[8][16][32];  // x features, f16, K-padded to 32
    __shared__ _Float16 hh_s[8][16][32];  // h state,    f16, K-padded to 32
    __shared__ float    xs_s[8][16][32];  // x @ pk  pre-activations
    __shared__ float    rs_s[8][16][32];  // h @ prk pre-activations
    __shared__ float    pb_s[60];
    __shared__ float    fsum[19];         // 4 m + 4 gamma + 1 nu_rel + 10 h_new

    const int tid  = threadIdx.x;
    const int wave = tid >> 5;
    const int lane = tid & 31;
    const int row  = lane & 15;
    const int hi   = lane >> 4;
    const long blockBase = (long)blockIdx.x * 128;

    if (tid < 19) fsum[tid] = 0.f;
    if (tid < 60) pb_s[tid] = pb[tid];
    __syncthreads();

    // ---- phase 1: lanes 0-15 of each wave compute + stage one element -----
    if (hi == 0) {
        const long e  = blockBase + wave * 16 + row;
        const long ec = (e < n) ? e : (long)(n - 1);
        const float scale   = (float)n * rsqrtf(ws[0]);   // n / ||d_theta||
        const float mean_lr = ws[1] / (float)n;

        float x[P_IN];
        float hvec[P_UNITS];
        float gb[4], lm[4], ll[4];
        float llm = 0.f;
#pragma unroll
        for (int s = 0; s < 4; ++s) {
            gb[s] = gbar_new[(size_t)s * n + ec];
            lm[s] = lam_new[(size_t)s * n + ec];
            ll[s] = logf(lm[s]);
            llm  += ll[s];
        }
        llm *= 0.25f;
#pragma unroll
        for (int s = 0; s < 4; ++s) {
            x[s]     = gb[s] * rsqrtf(lm[s]);   // m
            x[4 + s] = ll[s] - llm;             // gamma_feat
        }
        float lrv = lr[ec];
        x[8] = lrv - mean_lr;                   // nu_rel
#pragma unroll
        for (int t = 0; t < T_UNITS; ++t) x[9 + t]  = h_tensor[t];
#pragma unroll
        for (int t = 0; t < G_UNITS; ++t) x[14 + t] = h_global[t];
#pragma unroll
        for (int u = 0; u < P_UNITS; ++u) hvec[u] = h_param[ec * P_UNITS + u];

        // stage as f16 rows (contiguous vector stores -> ds_store_b128)
#pragma unroll
        for (int c = 0; c < 4; ++c) {
            v8h vx, vh;
#pragma unroll
            for (int j = 0; j < 8; ++j) {
                int K = c * 8 + j;
                vx[j] = (_Float16)((K < P_IN)    ? x[K]    : 0.f);
                vh[j] = (_Float16)((K < P_UNITS) ? hvec[K] : 0.f);
            }
            *(v8h*)&xh_s[wave][row][c * 8] = vx;
            *(v8h*)&hh_s[wave][row][c * 8] = vh;
        }

        if (e < n) {
            float dt = b_dt[0];
#pragma unroll
            for (int u = 0; u < P_UNITS; ++u) dt += hvec[u] * w_dt[u];
            o_delta[e] = expf(lrv) * dt * scale;
#pragma unroll
            for (int k = 0; k < 9; ++k) atomicAdd(&fsum[k], x[k]);
        }
    }
    __syncthreads();

    // ---- phase 2: assemble fragments and run WMMA --------------------------
    // A k order used here (16x32, same as B): lane holds row (lane&15); halves 0..15 are
    // K = 16*hi..16*hi+15 (contiguous -> two ds_load_b128 each).
    v8h axl = *(const v8h*)&xh_s[wave][row][16 * hi];
    v8h axh = *(const v8h*)&xh_s[wave][row][16 * hi + 8];
    v8h ahl = *(const v8h*)&hh_s[wave][row][16 * hi];
    v8h ahh = *(const v8h*)&hh_s[wave][row][16 * hi + 8];
    v16h ax = __builtin_shufflevector(axl, axh, 0,1,2,3,4,5,6,7,8,9,10,11,12,13,14,15);
    v16h ah = __builtin_shufflevector(ahl, ahh, 0,1,2,3,4,5,6,7,8,9,10,11,12,13,14,15);

    // B fragments: precomputed packed f16 in ws (128 B per lane)
    const char* fp = (const char*)(ws + WS_FRAG_OFF) + lane * 128;
    v8h b0l = *(const v8h*)(fp +   0), b0h = *(const v8h*)(fp +  16);
    v8h b1l = *(const v8h*)(fp +  32), b1h = *(const v8h*)(fp +  48);
    v8h b2l = *(const v8h*)(fp +  64), b2h = *(const v8h*)(fp +  80);
    v8h b3l = *(const v8h*)(fp +  96), b3h = *(const v8h*)(fp + 112);
    v16h bx_lo = __builtin_shufflevector(b0l, b0h, 0,1,2,3,4,5,6,7,8,9,10,11,12,13,14,15);
    v16h bx_hi = __builtin_shufflevector(b1l, b1h, 0,1,2,3,4,5,6,7,8,9,10,11,12,13,14,15);
    v16h bh_lo = __builtin_shufflevector(b2l, b2h, 0,1,2,3,4,5,6,7,8,9,10,11,12,13,14,15);
    v16h bh_hi = __builtin_shufflevector(b3l, b3h, 0,1,2,3,4,5,6,7,8,9,10,11,12,13,14,15);

    v8f c0 = {};
    v8f cxl = __builtin_amdgcn_wmma_f32_16x16x32_f16(false, ax, false, bx_lo, (short)0, c0, false, false);
    v8f cxh = __builtin_amdgcn_wmma_f32_16x16x32_f16(false, ax, false, bx_hi, (short)0, c0, false, false);
    v8f crl = __builtin_amdgcn_wmma_f32_16x16x32_f16(false, ah, false, bh_lo, (short)0, c0, false, false);
    v8f crh = __builtin_amdgcn_wmma_f32_16x16x32_f16(false, ah, false, bh_hi, (short)0, c0, false, false);

    // scatter D tiles to LDS per C/D layout: lane holds col (lane&15), rows i+8*hi
#pragma unroll
    for (int i = 0; i < 8; ++i) {
        int m = i + 8 * hi;
        xs_s[wave][m][row]      = cxl[i];
        xs_s[wave][m][16 + row] = cxh[i];
        rs_s[wave][m][row]      = crl[i];
        rs_s[wave][m][16 + row] = crh[i];
    }
    __syncthreads();

    // ---- phase 3: gate math: 128 elements x 10 units = 5 tasks/thread ------
#pragma unroll
    for (int rep = 0; rep < 5; ++rep) {
        int task = rep * 256 + tid;
        int el = task / 10;
        int u  = task - el * 10;
        long eg = blockBase + el;
        if (eg < n) {
            int w = el >> 4, r = el & 15;
            float xz = xs_s[w][r][u]      + pb_s[u];
            float xr = xs_s[w][r][10 + u] + pb_s[10 + u];
            float xh = xs_s[w][r][20 + u] + pb_s[20 + u];
            float rz = rs_s[w][r][u]      + pb_s[30 + u];
            float rr = rs_s[w][r][10 + u] + pb_s[40 + u];
            float rh = rs_s[w][r][20 + u] + pb_s[50 + u];
            float z  = 1.f / (1.f + expf(-(xz + rz)));
            float rg = 1.f / (1.f + expf(-(xr + rr)));
            float hh = tanhf(xh + rg * rh);
            float hold = h_param[eg * P_UNITS + u];
            float hnew = z * hold + (1.f - z) * hh;
            o_hparam[eg * P_UNITS + u] = hnew;
            atomicAdd(&fsum[9 + u], hnew);
        }
    }
    __syncthreads();
    if (tid < 19) atomicAdd(&ws[2 + tid], fsum[tid]);
}

// ---------------------------------------------------------------------------
// Final: tensor-level GRU (tiny) + global_inputs
// ---------------------------------------------------------------------------
__global__ void opt_final(
    const float* __restrict__ h_tensor, const float* __restrict__ h_global,
    const float* __restrict__ tk, const float* __restrict__ trk,
    const float* __restrict__ tb, const float* __restrict__ ws,
    float* __restrict__ o_ht, float* __restrict__ o_gi, int n)
{
    if (blockIdx.x != 0 || threadIdx.x != 0) return;
    float inv = 1.f / (float)n;
    float ti[T_IN];
#pragma unroll
    for (int k = 0; k < 9; ++k)       ti[k]      = ws[2 + k]  * inv;  // m_t, gamma_t, nu_t
#pragma unroll
    for (int u = 0; u < P_UNITS; ++u) ti[9 + u]  = ws[11 + u] * inv;  // hp_t
#pragma unroll
    for (int t = 0; t < G_UNITS; ++t) ti[19 + t] = h_global[t];

    float xs[15], rs[15];
#pragma unroll
    for (int j = 0; j < 15; ++j) {
        float a = tb[j];
        for (int k = 0; k < T_IN; ++k) a += ti[k] * tk[k * 15 + j];
        xs[j] = a;
        float b = tb[15 + j];
        for (int k = 0; k < T_UNITS; ++k) b += h_tensor[k] * trk[k * 15 + j];
        rs[j] = b;
    }
    float htn[T_UNITS];
#pragma unroll
    for (int u = 0; u < T_UNITS; ++u) {
        float z  = 1.f / (1.f + expf(-(xs[u] + rs[u])));
        float r  = 1.f / (1.f + expf(-(xs[5 + u] + rs[5 + u])));
        float hh = tanhf(xs[10 + u] + r * rs[10 + u]);
        htn[u] = z * h_tensor[u] + (1.f - z) * hh;
        o_ht[u] = htn[u];
    }
#pragma unroll
    for (int k = 0; k < 19; ++k) o_gi[k] = ti[k];      // m_t, gamma_t, nu_t, hp_t
#pragma unroll
    for (int u = 0; u < T_UNITS; ++u) o_gi[19 + u] = htn[u];
}

// ---------------------------------------------------------------------------
extern "C" void kernel_launch(void* const* d_in, const int* in_sizes, int n_in,
                              void* d_out, int out_size, void* d_ws, size_t ws_size,
                              hipStream_t stream)
{
    (void)n_in; (void)out_size; (void)ws_size;
    const float* grads    = (const float*)d_in[1];
    const float* g_bar    = (const float*)d_in[2];
    const float* lam      = (const float*)d_in[3];
    const float* h_param  = (const float*)d_in[4];
    const float* h_tensor = (const float*)d_in[5];
    const float* h_global = (const float*)d_in[6];
    const float* nu_bar   = (const float*)d_in[7];
    const float* lr       = (const float*)d_in[8];
    const float* pk       = (const float*)d_in[9];
    const float* prk      = (const float*)d_in[10];
    const float* pb       = (const float*)d_in[11];
    const float* tk       = (const float*)d_in[12];
    const float* trk      = (const float*)d_in[13];
    const float* tb       = (const float*)d_in[14];
    const float* w_dt     = (const float*)d_in[15];
    const float* b_dt     = (const float*)d_in[16];
    const float* w_dn     = (const float*)d_in[17];
    const float* b_dn     = (const float*)d_in[18];
    const float* w_bg     = (const float*)d_in[19];
    const float* b_bg     = (const float*)d_in[20];
    const float* w_bl     = (const float*)d_in[21];
    const float* b_bl     = (const float*)d_in[22];
    const float* gamma_p  = (const float*)d_in[23];

    const int n = in_sizes[0];   // N = 1024*2048

    float* out      = (float*)d_out;
    float* o_delta  = out;                       // [0, N)
    float* o_gbar   = out + (size_t)n;           // [N, 5N)
    float* o_lam    = out + (size_t)5 * n;       // [5N, 9N)
    float* o_hparam = out + (size_t)9 * n;       // [9N, 19N)
    float* o_ht     = out + (size_t)19 * n;      // [19N, 19N+5)
    float* o_nu     = o_ht + T_UNITS;            // [19N+5, 20N+5)
    float* o_nubar  = o_nu + (size_t)n;          // [20N+5, 21N+5)
    float* o_gi     = o_nubar + (size_t)n;       // [21N+5, 21N+29)
    float* ws       = (float*)d_ws;

    opt_init<<<1, 32, 0, stream>>>(ws);
    opt_prep<<<1, 32, 0, stream>>>(pk, prk, ws);

    opt_passA<<<(n + 255) / 256, 256, 0, stream>>>(
        grads, g_bar, lam, h_param, nu_bar, lr,
        w_dt, b_dt, w_dn, b_dn, w_bg, b_bg, w_bl, b_bl, gamma_p,
        o_gbar, o_lam, o_nu, o_nubar, ws, n);

    opt_passB<<<(n + 127) / 128, 256, 0, stream>>>(
        o_gbar, o_lam, h_param, lr, h_tensor, h_global,
        pb, w_dt, b_dt, o_delta, o_hparam, ws, n);

    opt_final<<<1, 32, 0, stream>>>(h_tensor, h_global, tk, trk, tb, ws, o_ht, o_gi, n);
}
